// ExampleNet_32384053412433
// MI455X (gfx1250) — matrix operand order repaired and checked by reading, not compiled, not run on hardware
//
#include <hip/hip_runtime.h>
#include <hip/hip_bf16.h>

typedef long long ll;
typedef __attribute__((ext_vector_type(16))) __bf16 v16bf;
typedef __attribute__((ext_vector_type(8)))  float  v8f;

union Vu { v16bf v; unsigned u[8]; };

__device__ __forceinline__ unsigned short f2bf(float f) {
    union { float f; unsigned u; } x; x.f = f;
    unsigned r = x.u + 0x7FFFu + ((x.u >> 16) & 1u);
    return (unsigned short)(r >> 16);
}
__device__ __forceinline__ float bf2f(unsigned short h) {
    union { unsigned u; float f; } x; x.u = ((unsigned)h) << 16;
    return x.f;
}
__device__ __forceinline__ float sigmoidf_(float x) { return 1.0f / (1.0f + expf(-x)); }

// ---------------------------------------------------------------------------
// Weight convert + zero-pad:  f32 [rows,K] -> bf16 [rows,Kp]
// ---------------------------------------------------------------------------
__global__ void padW_kernel(const float* __restrict__ src, unsigned short* __restrict__ dst,
                            int rows, int K, int Kp) {
    ll total = (ll)rows * Kp;
    for (ll i = (ll)blockIdx.x * blockDim.x + threadIdx.x; i < total;
         i += (ll)gridDim.x * blockDim.x) {
        ll r = i / Kp; int k = (int)(i - r * Kp);
        dst[i] = (k < K) ? f2bf(src[r * K + k]) : (unsigned short)0;
    }
}

// ---------------------------------------------------------------------------
// Row L2 norms of a bf16 matrix [rows, cols]
// ---------------------------------------------------------------------------
__global__ void rowNorm_kernel(const unsigned short* __restrict__ x, float* __restrict__ out,
                               ll rows, int cols) {
    ll r = (ll)blockIdx.x * blockDim.x + threadIdx.x;
    if (r >= rows) return;
    const unsigned short* p = x + r * cols;
    float s = 0.f;
    for (int j = 0; j < cols; ++j) { float v = bf2f(p[j]); s += v * v; }
    out[r] = sqrtf(s);
}

// ---------------------------------------------------------------------------
// Fused bidirectional-GRU direction kernel (persistent over the time loop).
// ---------------------------------------------------------------------------
struct GruP {
    const void* X; int xIsF32;
    const unsigned short* Wih;   // [2, G, Ep] bf16 (padded)
    const unsigned short* Whh;   // [2, G, H]  bf16
    const float* bih;            // [2, G]
    const float* bhh;            // [2, G]
    unsigned short* Y;           // [rows, T, 2H] bf16 or null
    float* Hout;                 // [rows, 2H] f32 or null
    int rows, T, E, Ep, H, G;
};

__device__ __forceinline__ void gru_loadAB(const unsigned short* __restrict__ src,
                                           const unsigned short* __restrict__ Wrow,
                                           int Kp, int k0, int mlm, int half,
                                           Vu& a, Vu& b) {
#pragma unroll
    for (int v = 0; v < 8; ++v) {
        int ka = k0 + (half << 4) + (v << 1);
        a.u[v] = *(const unsigned*)(src + mlm * Kp + ka);
        int kb = k0 + (half << 4) + (v << 1);
        b.u[v] = *(const unsigned*)(Wrow + kb);
    }
}

__global__ void gru_kernel(GruP p) {
    extern __shared__ char smraw[];
    const int H = p.H, G = p.G, Ep = p.Ep;
    float*          hF = (float*)smraw;                 // 16*H f32
    float*          gh = hF + 16 * H;                   // 16*G f32
    float*          gx = gh + 16 * G;                   // 16*G f32
    unsigned short* hB = (unsigned short*)(gx + 16 * G);// 16*H bf16
    unsigned short* xB = hB + 16 * H;                   // 16*Ep bf16

    const int dir = blockIdx.y;
    const int rowBase = blockIdx.x * 16;
    const unsigned short* Wih = p.Wih + (ll)dir * G * Ep;
    const unsigned short* Whh = p.Whh + (ll)dir * G * H;
    const float* bih = p.bih + (ll)dir * G;
    const float* bhh = p.bhh + (ll)dir * G;

    const int tid  = threadIdx.x;
    const int lane = tid & 31;
    const int wave = tid >> 5;
    const int nw   = blockDim.x >> 5;
    const int mlm  = lane & 15;   // m (A) / n (B/C) within tile
    const int half = lane >> 4;
    const int nTiles = G >> 4;

    for (int i = tid; i < 16 * H; i += blockDim.x) { hF[i] = 0.f; hB[i] = 0; }
    __syncthreads();

    for (int step = 0; step < p.T; ++step) {
        const int tt = dir ? (p.T - 1 - step) : step;

        // stage x_t into LDS (convert f32 -> bf16 if needed; zero the K pad)
        for (int i = tid; i < 16 * Ep; i += blockDim.x) {
            int m = i / Ep, k = i - m * Ep;
            unsigned short v = 0;
            if (k < p.E) {
                ll off = ((ll)(rowBase + m) * p.T + tt) * p.E + k;
                v = p.xIsF32 ? f2bf(((const float*)p.X)[off])
                             : ((const unsigned short*)p.X)[off];
            }
            xB[i] = v;
        }
        __syncthreads();

        // two WMMA passes: pass 0 -> gh (K=H), pass 1 -> gx (K=Ep)
        // all K depths here are multiples of 64 -> clean 2x-unrolled loop
        for (int pass = 0; pass < 2; ++pass) {
            const unsigned short* src = pass ? xB : hB;
            const int Kp              = pass ? Ep : H;
            const unsigned short* W   = pass ? Wih : Whh;
            float* dst                = pass ? gx : gh;
            for (int nt = wave; nt < nTiles; nt += nw) {
                const unsigned short* Wrow = W + (ll)(nt * 16 + mlm) * Kp;
                if (nt + nw < nTiles)
                    __builtin_prefetch(W + (ll)((nt + nw) * 16 + mlm) * Kp, 0, 3);
                v8f acc = {0.f, 0.f, 0.f, 0.f, 0.f, 0.f, 0.f, 0.f};
                for (int k0 = 0; k0 < Kp; k0 += 64) {
                    Vu a0, b0, a1, b1;
                    gru_loadAB(src, Wrow, Kp, k0,      mlm, half, a0, b0);
                    gru_loadAB(src, Wrow, Kp, k0 + 32, mlm, half, a1, b1);
                    acc = __builtin_amdgcn_wmma_f32_16x16x32_bf16(
                        false, a0.v, false, b0.v, (short)0, acc, false, false);
                    acc = __builtin_amdgcn_wmma_f32_16x16x32_bf16(
                        false, a1.v, false, b1.v, (short)0, acc, false, false);
                }
#pragma unroll
                for (int v = 0; v < 8; ++v)
                    dst[(v + half * 8) * G + nt * 16 + mlm] = acc[v];
            }
        }
        __syncthreads();

        // GRU cell update
        for (int i = tid; i < 16 * H; i += blockDim.x) {
            int m = i / H, j = i - m * H;
            float r = sigmoidf_(gx[m * G + j]     + bih[j]     + gh[m * G + j]     + bhh[j]);
            float z = sigmoidf_(gx[m * G + H + j] + bih[H + j] + gh[m * G + H + j] + bhh[H + j]);
            float nn = tanhf(gx[m * G + 2 * H + j] + bih[2 * H + j] +
                             r * (gh[m * G + 2 * H + j] + bhh[2 * H + j]));
            float hn = (1.f - z) * nn + z * hF[i];
            hF[i] = hn;
            hB[i] = f2bf(hn);
            if (p.Y)
                p.Y[(ll)(rowBase + m) * p.T * 2 * H + (ll)tt * 2 * H + dir * H + j] = f2bf(hn);
            if (p.Hout && step == p.T - 1)
                p.Hout[(ll)(rowBase + m) * 2 * H + dir * H + j] = hn;
        }
        __syncthreads();
    }
}

// ---------------------------------------------------------------------------
// Generic strided bf16 WMMA GEMM over 640 (b,n) pairs, one 16x16 C tile/wave.
// ---------------------------------------------------------------------------
struct GemmP {
    const unsigned short* A; ll aPairStride; int aPairDiv; int Asm, Ask;
    const unsigned short* Bm; ll bPairStride; int bPairDiv; int Bsk, Bsn;
    unsigned short* C; ll cPairStride;
    int M, N, K;
    const float* ncB; int ncDiv, ncStride;   // epilogue (null = off)
    const float* noB; int noStride;
};

__device__ __forceinline__ void gemm_loadA(const GemmP& p, const unsigned short* __restrict__ A,
                                           int m, int k0, int half, Vu& a) {
    const bool mok = m < p.M;
    const unsigned short* Arow = A + (ll)m * p.Asm;
#pragma unroll
    for (int v = 0; v < 8; ++v) {
        int ka = k0 + (half << 4) + (v << 1);
        unsigned val;
        if (p.Ask == 1) {           // contiguous-K fast path: one b32 load
            if (mok && ka + 1 < p.K)      val = *(const unsigned*)(Arow + ka);
            else if (mok && ka < p.K)     val = (unsigned)Arow[ka];
            else                          val = 0u;
        } else {
            unsigned lo = (mok && ka     < p.K) ? Arow[(ll)ka * p.Ask]       : 0u;
            unsigned hi = (mok && ka + 1 < p.K) ? Arow[(ll)(ka + 1) * p.Ask] : 0u;
            val = lo | (hi << 16);
        }
        a.u[v] = val;
    }
}

__device__ __forceinline__ void gemm_loadB(const GemmP& p, const unsigned short* __restrict__ B,
                                           int n, int k0, int half, Vu& b) {
    const bool nok = n < p.N;
    const unsigned short* Bcol = B + (ll)n * p.Bsn;
#pragma unroll
    for (int v = 0; v < 8; ++v) {
        int kb = k0 + (half << 4) + (v << 1);
        unsigned val;
        if (p.Bsk == 1) {           // contiguous-K fast path: one b32 load
            if (nok && kb + 1 < p.K)      val = *(const unsigned*)(Bcol + kb);
            else if (nok && kb < p.K)     val = (unsigned)Bcol[kb];
            else                          val = 0u;
        } else {
            unsigned lo = (nok && kb     < p.K) ? Bcol[(ll)kb * p.Bsk]       : 0u;
            unsigned hi = (nok && kb + 1 < p.K) ? Bcol[(ll)(kb + 1) * p.Bsk] : 0u;
            val = lo | (hi << 16);
        }
        b.u[v] = val;
    }
}

__global__ void gemm_bf16_kernel(GemmP p) {
    const int pair = blockIdx.x;
    const unsigned short* A = p.A  + (ll)(pair / p.aPairDiv) * p.aPairStride;
    const unsigned short* B = p.Bm + (ll)(pair / p.bPairDiv) * p.bPairStride;
    unsigned short* C = p.C + (ll)pair * p.cPairStride;

    const int lane = threadIdx.x & 31;
    const int wave = threadIdx.x >> 5;
    const int nw   = blockDim.x >> 5;
    const int mlm  = lane & 15;
    const int half = lane >> 4;

    const int Mt = (p.M + 15) >> 4, Nt = (p.N + 15) >> 4;
    const int total = Mt * Nt;

    for (int tix = blockIdx.y * nw + wave; tix < total; tix += gridDim.y * nw) {
        const int mt = tix / Nt, nt = tix - mt * Nt;
        const int m = mt * 16 + mlm;
        const int n = nt * 16 + mlm;
        v8f acc = {0.f, 0.f, 0.f, 0.f, 0.f, 0.f, 0.f, 0.f};
        int k0 = 0;
        // bulk: 2x-unrolled, two load clauses in flight per pair of WMMAs
        for (; k0 + 64 <= p.K; k0 += 64) {
            Vu a0, b0, a1, b1;
            gemm_loadA(p, A, m, k0,      half, a0);
            gemm_loadB(p, B, n, k0,      half, b0);
            gemm_loadA(p, A, m, k0 + 32, half, a1);
            gemm_loadB(p, B, n, k0 + 32, half, b1);
            acc = __builtin_amdgcn_wmma_f32_16x16x32_bf16(
                false, a0.v, false, b0.v, (short)0, acc, false, false);
            acc = __builtin_amdgcn_wmma_f32_16x16x32_bf16(
                false, a1.v, false, b1.v, (short)0, acc, false, false);
        }
        // remainder: up to two masked 32-wide steps
        for (; k0 < p.K; k0 += 32) {
            Vu a0, b0;
            gemm_loadA(p, A, m, k0, half, a0);
            gemm_loadB(p, B, n, k0, half, b0);
            acc = __builtin_amdgcn_wmma_f32_16x16x32_bf16(
                false, a0.v, false, b0.v, (short)0, acc, false, false);
        }
        const float* ncp = p.ncB ? p.ncB + (ll)(pair / p.ncDiv) * p.ncStride : nullptr;
        const float* nop = p.ncB ? p.noB + (ll)pair * p.noStride : nullptr;
#pragma unroll
        for (int v = 0; v < 8; ++v) {
            int mm = mt * 16 + v + half * 8;
            if (mm < p.M && n < p.N) {
                float val = acc[v];
                if (ncp) val = val / fmaxf(ncp[mm] * nop[n], 1e-8f);
                C[(ll)mm * p.N + n] = f2bf(val);
            }
        }
    }
}

// ---------------------------------------------------------------------------
// Final: logits = cos(hCon, hOp) over 512 dims; softmax over N=10 per batch.
// ---------------------------------------------------------------------------
__global__ void cosSoftmax_kernel(const float* __restrict__ hc, const float* __restrict__ ho,
                                  float* __restrict__ out) {
    __shared__ float lg[16];
    const int b = blockIdx.x, t = threadIdx.x;
    if (t < 10) {
        const float* a = hc + ((ll)b * 10 + t) * 512;
        const float* c = ho + ((ll)b * 10 + t) * 512;
        float num = 0.f, na = 0.f, nb = 0.f;
        for (int j = 0; j < 512; ++j) { num += a[j] * c[j]; na += a[j] * a[j]; nb += c[j] * c[j]; }
        lg[t] = num / fmaxf(sqrtf(na) * sqrtf(nb), 1e-8f);
    }
    __syncthreads();
    if (t == 0) {
        float mx = -1e30f;
        for (int i = 0; i < 10; ++i) mx = fmaxf(mx, lg[i]);
        float e[10], s = 0.f;
        for (int i = 0; i < 10; ++i) { e[i] = expf(lg[i] - mx); s += e[i]; }
        for (int i = 0; i < 10; ++i) out[(ll)b * 10 + i] = e[i] / s;
    }
}

// ---------------------------------------------------------------------------
extern "C" void kernel_launch(void* const* d_in, const int* in_sizes, int n_in,
                              void* d_out, int out_size, void* d_ws, size_t ws_size,
                              hipStream_t stream) {
    (void)in_sizes; (void)n_in; (void)out_size; (void)ws_size;
    const float* context = (const float*)d_in[0];   // [64,400,300]
    const float* options = (const float*)d_in[2];   // [64,10,50,300]
    const float* W1ih = (const float*)d_in[4];      // [2,384,300]
    const float* W1hh = (const float*)d_in[5];      // [2,384,128]
    const float* b1ih = (const float*)d_in[6];
    const float* b1hh = (const float*)d_in[7];
    const float* W2ih = (const float*)d_in[8];      // [2,768,256]
    const float* W2hh = (const float*)d_in[9];
    const float* b2ih = (const float*)d_in[10];
    const float* b2hh = (const float*)d_in[11];

    // workspace carve-out
    char* ws = (char*)d_ws;
    size_t off = 0;
    auto carve = [&](size_t bytes) -> void* {
        void* p = ws + off;
        off = (off + bytes + 255) & ~(size_t)255;
        return p;
    };
    unsigned short* wih1p = (unsigned short*)carve((size_t)2 * 384 * 320 * 2);
    unsigned short* whh1p = (unsigned short*)carve((size_t)2 * 384 * 128 * 2);
    unsigned short* wih2p = (unsigned short*)carve((size_t)2 * 768 * 256 * 2);
    unsigned short* whh2p = (unsigned short*)carve((size_t)2 * 768 * 256 * 2);
    unsigned short* ctxO  = (unsigned short*)carve((size_t)64 * 400 * 256 * 2);
    unsigned short* optO  = (unsigned short*)carve((size_t)640 * 50 * 256 * 2);
    float*          ncv   = (float*)carve((size_t)25600 * 4);
    float*          nov   = (float*)carve((size_t)32000 * 4);
    unsigned short* simB  = (unsigned short*)carve((size_t)640 * 400 * 50 * 2);
    unsigned short* aCon  = (unsigned short*)carve((size_t)640 * 50 * 256 * 2);
    unsigned short* aOp   = (unsigned short*)carve((size_t)640 * 400 * 256 * 2);
    float*          hCon  = (float*)carve((size_t)640 * 512 * 4);
    float*          hOp   = (float*)carve((size_t)640 * 512 * 4);

    // 1) weight conversion (f32 -> bf16, K-padded)
    padW_kernel<<<960, 256, 0, stream>>>(W1ih, wih1p, 2 * 384, 300, 320);
    padW_kernel<<<384, 256, 0, stream>>>(W1hh, whh1p, 2 * 384, 128, 128);
    padW_kernel<<<1536, 256, 0, stream>>>(W2ih, wih2p, 2 * 768, 256, 256);
    padW_kernel<<<1536, 256, 0, stream>>>(W2hh, whh2p, 2 * 768, 256, 256);

    // 2) rnn1 on context and options (fused time loop, WMMA recurrence)
    {
        GruP g{}; g.X = context; g.xIsF32 = 1; g.Wih = wih1p; g.Whh = whh1p;
        g.bih = b1ih; g.bhh = b1hh; g.Y = ctxO; g.Hout = nullptr;
        g.rows = 64; g.T = 400; g.E = 300; g.Ep = 320; g.H = 128; g.G = 384;
        size_t sh = (size_t)16 * 128 * 4 + 2 * (size_t)16 * 384 * 4 + 16 * 128 * 2 + 16 * 320 * 2;
        gru_kernel<<<dim3(4, 2), 512, sh, stream>>>(g);
        g.X = options; g.Y = optO; g.rows = 640; g.T = 50;
        gru_kernel<<<dim3(40, 2), 512, sh, stream>>>(g);
    }

    // 3) row norms for cosine attention
    rowNorm_kernel<<<100, 256, 0, stream>>>(ctxO, ncv, 25600, 256);
    rowNorm_kernel<<<125, 256, 0, stream>>>(optO, nov, 32000, 256);

    // 4) sim[b,n] = (ctx_out[b] @ opt_out[b,n]^T) / (nc*no)   [400,50] per pair
    {
        GemmP g{};
        g.A = ctxO; g.aPairStride = (ll)400 * 256; g.aPairDiv = 10; g.Asm = 256; g.Ask = 1;
        g.Bm = optO; g.bPairStride = (ll)50 * 256; g.bPairDiv = 1;  g.Bsk = 1;  g.Bsn = 256;
        g.C = simB; g.cPairStride = (ll)400 * 50;
        g.M = 400; g.N = 50; g.K = 256;
        g.ncB = ncv; g.ncDiv = 10; g.ncStride = 400; g.noB = nov; g.noStride = 50;
        gemm_bf16_kernel<<<dim3(640, 13), 256, 0, stream>>>(g);
    }
    // 5) attenCon[b,n] = sim[b,n]^T @ ctx_out[b]   [50,256]
    {
        GemmP g{};
        g.A = simB; g.aPairStride = (ll)400 * 50; g.aPairDiv = 1; g.Asm = 1; g.Ask = 50;
        g.Bm = ctxO; g.bPairStride = (ll)400 * 256; g.bPairDiv = 10; g.Bsk = 256; g.Bsn = 1;
        g.C = aCon; g.cPairStride = (ll)50 * 256;
        g.M = 50; g.N = 256; g.K = 400;
        g.ncB = nullptr; g.noB = nullptr; g.ncDiv = 1; g.ncStride = 0; g.noStride = 0;
        gemm_bf16_kernel<<<dim3(640, 8), 256, 0, stream>>>(g);
    }
    // 6) attenOp[b,n] = sim[b,n] @ opt_out[b,n]    [400,256]
    {
        GemmP g{};
        g.A = simB; g.aPairStride = (ll)400 * 50; g.aPairDiv = 1; g.Asm = 50; g.Ask = 1;
        g.Bm = optO; g.bPairStride = (ll)50 * 256; g.bPairDiv = 1; g.Bsk = 256; g.Bsn = 1;
        g.C = aOp; g.cPairStride = (ll)400 * 256;
        g.M = 400; g.N = 256; g.K = 50;
        g.ncB = nullptr; g.noB = nullptr; g.ncDiv = 1; g.ncStride = 0; g.noStride = 0;
        gemm_bf16_kernel<<<dim3(640, 50), 256, 0, stream>>>(g);
    }

    // 7) rnn2 on attenCon (T=50) and attenOp (T=400); keep only final hidden
    {
        GruP g{}; g.X = aCon; g.xIsF32 = 0; g.Wih = wih2p; g.Whh = whh2p;
        g.bih = b2ih; g.bhh = b2hh; g.Y = nullptr; g.Hout = hCon;
        g.rows = 640; g.T = 50; g.E = 256; g.Ep = 256; g.H = 256; g.G = 768;
        size_t sh = (size_t)16 * 256 * 4 + 2 * (size_t)16 * 768 * 4 + 16 * 256 * 2 + 16 * 256 * 2;
        gru_kernel<<<dim3(40, 2), 512, sh, stream>>>(g);
        g.X = aOp; g.Hout = hOp; g.T = 400;
        gru_kernel<<<dim3(40, 2), 512, sh, stream>>>(g);
    }

    // 8) cosine similarity of final hiddens + softmax over N=10
    cosSoftmax_kernel<<<64, 32, 0, stream>>>(hCon, hOp, (float*)d_out);
}
